// ContextNet_17815524344264
// MI455X (gfx1250) — matrix operand order repaired and checked by reading, not compiled, not run on hardware
//
#include <hip/hip_runtime.h>
#include <hip/hip_bf16.h>
#include <math.h>

// ---------------------------------------------------------------------------
// CDNA5 (gfx1250) WMMA types
// ---------------------------------------------------------------------------
typedef __attribute__((ext_vector_type(16))) __bf16 v16bf;
typedef __attribute__((ext_vector_type(8)))  __bf16 v8bf;
typedef __attribute__((ext_vector_type(8)))  float  v8f;

#define NPTS   16384
#define BATCH  2
#define BNPTS  (BATCH * NPTS)
#define EMB    256
#define COND   16
#define CTX    64
// eval-mode BatchNorm with running stats (0,1): scale = g / sqrt(1 + 1e-4)
#define BNINV  0.999950003749688f

static __device__ __forceinline__ float sigmoidf_(float x) { return 1.f / (1.f + expf(-x)); }

// ---------------------------------------------------------------------------
// Embedding: emb = silu(temb(t)@Wt+bt) + silu(cond@Wc+bc); ctx_glb = emb@Wce+bce
// ---------------------------------------------------------------------------
__global__ void emb_kernel(const float* __restrict__ t, const float* __restrict__ cond,
                           const float* __restrict__ wt, const float* __restrict__ bt,
                           const float* __restrict__ wc, const float* __restrict__ bc,
                           const float* __restrict__ wce, const float* __restrict__ bce,
                           float* __restrict__ emb, float* __restrict__ ctxg) {
    int b = blockIdx.x;
    int o = threadIdx.x;
    __shared__ float e[EMB];
    __shared__ float em[EMB];
    if (o < 128) {
        float fr = expf(-9.210340371976184f * (float)o / 128.f);
        float a = t[b] * fr;
        e[o] = cosf(a);
        e[o + 128] = sinf(a);
    }
    __syncthreads();
    float s1 = bt[o];
    for (int i = 0; i < EMB; i++) s1 += e[i] * wt[i * EMB + o];
    s1 = s1 * sigmoidf_(s1);
    float s2 = bc[o];
    for (int i = 0; i < COND; i++) s2 += cond[b * COND + i] * wc[i * EMB + o];
    s2 = s2 * sigmoidf_(s2);
    float ev = s1 + s2;
    emb[b * EMB + o] = ev;
    em[o] = ev;
    __syncthreads();
    if (o < CTX) {
        float s = bce[o];
        for (int i = 0; i < EMB; i++) s += em[i] * wce[i * CTX + o];
        ctxg[b * CTX + o] = s;
    }
}

// ---------------------------------------------------------------------------
// Coordinate normalization (unit-sphere): mean, radius, ncu = nc/(2r+eps)+0.5
// ---------------------------------------------------------------------------
__global__ void coord_mean_kernel(const float* __restrict__ x, float* __restrict__ ctr) {
    int b = blockIdx.x;
    float s0 = 0.f, s1 = 0.f, s2 = 0.f;
    for (int n = threadIdx.x; n < NPTS; n += blockDim.x) {
        const float* p = x + ((size_t)b * NPTS + n) * 3;
        s0 += p[0]; s1 += p[1]; s2 += p[2];
    }
    __shared__ float r0[256], r1[256], r2[256];
    r0[threadIdx.x] = s0; r1[threadIdx.x] = s1; r2[threadIdx.x] = s2;
    __syncthreads();
    for (int st = 128; st > 0; st >>= 1) {
        if (threadIdx.x < st) {
            r0[threadIdx.x] += r0[threadIdx.x + st];
            r1[threadIdx.x] += r1[threadIdx.x + st];
            r2[threadIdx.x] += r2[threadIdx.x + st];
        }
        __syncthreads();
    }
    if (threadIdx.x == 0) {
        ctr[b * 3 + 0] = r0[0] / (float)NPTS;
        ctr[b * 3 + 1] = r1[0] / (float)NPTS;
        ctr[b * 3 + 2] = r2[0] / (float)NPTS;
    }
}

__global__ void coord_rad_kernel(const float* __restrict__ x, const float* __restrict__ ctr,
                                 float* __restrict__ rad) {
    int b = blockIdx.x;
    float c0 = ctr[b * 3 + 0], c1 = ctr[b * 3 + 1], c2 = ctr[b * 3 + 2];
    float mx = 0.f;
    for (int n = threadIdx.x; n < NPTS; n += blockDim.x) {
        const float* p = x + ((size_t)b * NPTS + n) * 3;
        float d0 = p[0] - c0, d1 = p[1] - c1, d2 = p[2] - c2;
        mx = fmaxf(mx, sqrtf(d0 * d0 + d1 * d1 + d2 * d2));
    }
    __shared__ float r[256];
    r[threadIdx.x] = mx;
    __syncthreads();
    for (int st = 128; st > 0; st >>= 1) {
        if (threadIdx.x < st) r[threadIdx.x] = fmaxf(r[threadIdx.x], r[threadIdx.x + st]);
        __syncthreads();
    }
    if (threadIdx.x == 0) rad[b] = r[0];
}

__global__ void ncu_kernel(const float* __restrict__ x, const float* __restrict__ ctr,
                           const float* __restrict__ rad, float* __restrict__ ncu) {
    size_t i = (size_t)blockIdx.x * blockDim.x + threadIdx.x;
    if (i >= (size_t)BNPTS) return;
    int b = (int)(i / NPTS);
    float rr = rad[b] * 2.f + 1e-6f;
    const float* p = x + i * 3;
    float* q = ncu + i * 3;
    for (int k = 0; k < 3; k++) q[k] = (p[k] - ctr[b * 3 + k]) / rr + 0.5f;
}

// ---------------------------------------------------------------------------
// Misc conversion / init kernels
// ---------------------------------------------------------------------------
__global__ void zero_f32(float* __restrict__ p, size_t n) {
    size_t i = (size_t)blockIdx.x * blockDim.x + threadIdx.x;
    if (i < n) p[i] = 0.f;
}

__global__ void build_f0_kernel(const float* __restrict__ emb, const float* __restrict__ x,
                                __bf16* __restrict__ xb) {
    size_t i = (size_t)blockIdx.x * blockDim.x + threadIdx.x;
    size_t total = (size_t)BNPTS * 288;
    if (i >= total) return;
    int c = (int)(i % 288);
    size_t pn = i / 288;
    int b = (int)(pn / NPTS);
    float v;
    if (c < EMB)      v = emb[b * EMB + c];
    else if (c < 259) v = x[pn * 3 + (c - 256)];
    else              v = 0.f;
    xb[i] = (__bf16)v;
}

__global__ void to_bf16_kernel(const float* __restrict__ f, __bf16* __restrict__ xb,
                               int C, int Cpad, size_t rows) {
    size_t i = (size_t)blockIdx.x * blockDim.x + threadIdx.x;
    size_t total = rows * (size_t)Cpad;
    if (i >= total) return;
    int c = (int)(i % Cpad);
    size_t r = i / Cpad;
    xb[i] = (__bf16)((c < C) ? f[r * (size_t)C + c] : 0.f);
}

// A-fragment layout within each 32-K block: store order [0..31] (contiguous)
// so lane (m, half) loads one contiguous v16bf at (k0 + half*16).
static __device__ __forceinline__ int aswz(int p) {
    int jj = p & 31;
    int oct = jj >> 3;
    int kk = (oct == 1) ? jj : (oct == 2) ? jj : jj;
    return (p & ~31) + kk;
}

__global__ void wconv_kernel(const float* __restrict__ w, __bf16* __restrict__ wb,
                             int O, int I, int Ipad) {
    size_t i = (size_t)blockIdx.x * blockDim.x + threadIdx.x;
    size_t total = (size_t)O * Ipad;
    if (i >= total) return;
    int p = (int)(i % Ipad);
    int o = (int)(i / Ipad);
    int K = aswz(p);
    wb[i] = (__bf16)((K < I) ? w[(size_t)o * I + K] : 0.f);
}

// (O,I,3,3,3) fp32  ->  bf16 [27][O][C] tap-major A matrices (K-swizzled)
__global__ void wconv3d_kernel(const float* __restrict__ w, __bf16* __restrict__ wb,
                               int O, int C) {
    size_t i = (size_t)blockIdx.x * blockDim.x + threadIdx.x;
    size_t total = (size_t)27 * O * C;
    if (i >= total) return;
    int cpos = (int)(i % C);
    size_t rest = i / C;
    int o = (int)(rest % O);
    int t = (int)(rest / O);
    int K = aswz(cpos);
    wb[i] = (__bf16)w[((size_t)o * C + K) * 27 + t];
}

// ---------------------------------------------------------------------------
// LDS staging helpers (B panels shared by all 4 waves of a block)
// ---------------------------------------------------------------------------
// Stage 64 columns x 32 K of X (bf16) into LDS: [64][32], 4KB, b128 chunks.
static __device__ __forceinline__ void stage_x(const __bf16* __restrict__ Xb, __bf16* dst,
                                               int n0, int Ipad, int k0, int tid) {
#pragma unroll
    for (int j = 0; j < 2; j++) {
        int chunk = tid * 2 + j;        // 0..255
        int row = chunk >> 2;           // column index 0..63
        int o16 = chunk & 3;            // 16B chunk within the 64B row
        v8bf v = *(const v8bf*)(Xb + (size_t)(n0 + row) * Ipad + k0 + o16 * 8);
        *(v8bf*)(dst + row * 32 + o16 * 8) = v;
    }
}

// Stage 16 neighbor-voxel columns x C channels for one conv tap into LDS: [16][C].
// Boundary columns are zero-filled here (keeps EXEC masking out of the WMMA loop).
static __device__ __forceinline__ void stage_g(const __bf16* __restrict__ G, __bf16* dst,
                                               int b, int r, int r3, int C, int t,
                                               int vtile, int tid) {
    int cpr = C >> 3;                    // 16B chunks per row
    int nch = 16 * cpr;
    int dz = t % 3 - 1, dy = (t / 3) % 3 - 1, dx = t / 9 - 1;
    for (int chunk = tid; chunk < nch; chunk += 128) {
        int rowc = chunk / cpr;          // voxel column 0..15
        int o16 = chunk % cpr;
        int vb = vtile * 16 + rowc;
        int vz = vb % r, vy = (vb / r) % r, vx = vb / (r * r);
        int nx = vx + dx, ny = vy + dy, nz = vz + dz;
        bool valid = (unsigned)nx < (unsigned)r && (unsigned)ny < (unsigned)r &&
                     (unsigned)nz < (unsigned)r;
        v8bf v;
        if (valid) {
            int nid = (nx * r + ny) * r + nz;
            v = *(const v8bf*)(G + ((size_t)b * r3 + nid) * C + o16 * 8);
        } else {
#pragma unroll
            for (int e = 0; e < 8; e++) v[e] = (__bf16)0.0f;
        }
        *(v8bf*)(dst + rowc * C + o16 * 8) = v;
    }
}

// ---------------------------------------------------------------------------
// WMMA GEMM:  Y[n][m] = act(bn(W[m][k]·X[n][k] + bias)) (+ res)
// X bf16 [Ncols][Ipad] point-major, W bf16 [O][Ipad] (K-swizzled), Y fp32 [Ncols][O]
// block = 128 threads (4 waves); wave -> 16 output channels; block -> 64 cols.
// B panel double-buffered through LDS (all 4 waves share it).
// grid = (Ncols/64, O/64)
// ---------------------------------------------------------------------------
template <int ACT, int HASBN, int HASRES>
__global__ __launch_bounds__(128) void gemm_wmma(const __bf16* __restrict__ Xb,
                                                 const __bf16* __restrict__ Wb,
                                                 const float* __restrict__ bias,
                                                 const float* __restrict__ bng,
                                                 const float* __restrict__ bnb,
                                                 const float* __restrict__ res,
                                                 float* __restrict__ Y,
                                                 int O, int Ipad) {
    __shared__ __align__(32) __bf16 sx[2][64 * 32];
    const int tid = threadIdx.x;
    const int lane = tid & 31;
    const int wv = tid >> 5;
    const int m0w = blockIdx.y * 64 + wv * 16;
    const int n0 = blockIdx.x * 64;
    const int col = lane & 15;
    const int half = lane >> 4;

    v8f acc[4];
#pragma unroll
    for (int j = 0; j < 4; j++)
#pragma unroll
        for (int v = 0; v < 8; v++) acc[j][v] = 0.f;

    const __bf16* wrow = Wb + (size_t)(m0w + col) * Ipad;
    __builtin_prefetch(wrow, 0, 2);

    stage_x(Xb, sx[0], n0, Ipad, 0, tid);
    __syncthreads();
    int p = 0;
    for (int k0 = 0; k0 < Ipad; k0 += 32) {
        if (k0 + 32 < Ipad) stage_x(Xb, sx[p ^ 1], n0, Ipad, k0 + 32, tid);
        v16bf a = *(const v16bf*)(wrow + k0 + half * 16);   // swizzled: single load
#pragma unroll
        for (int j = 0; j < 4; j++) {
            v16bf bf = *(const v16bf*)(sx[p] + (j * 16 + col) * 32 + half * 16);
            acc[j] = __builtin_amdgcn_wmma_f32_16x16x32_bf16(
                false, a, false, bf, (short)0, acc[j], false, false);
        }
        __syncthreads();
        p ^= 1;
    }

#pragma unroll
    for (int j = 0; j < 4; j++) {
        int n = n0 + j * 16 + col;
        float* yr = Y + (size_t)n * O + m0w + half * 8;
#pragma unroll
        for (int v = 0; v < 8; v++) {
            int m = m0w + half * 8 + v;
            float val = acc[j][v] + bias[m];
            if (HASBN) val = val * (bng[m] * BNINV) + bnb[m];
            if (ACT == 1) val = fmaxf(val, 0.f);
            if (HASRES) val += res[(size_t)n * O + m];
            yr[v] = val;
        }
    }
}

// ---------------------------------------------------------------------------
// conv3d 3x3x3 SAME as implicit GEMM over 27 taps, B panel via LDS per tap.
// G bf16 [B][r3][C] voxel-major, W27 bf16 [27][O][C] (K-swizzled), Yv fp32 [B][r3][C].
// fused eval-BN + leaky-relu(0.1) epilogue.
// grid = (r3/16, O/64, B), block = 128
// ---------------------------------------------------------------------------
__global__ __launch_bounds__(128) void conv3d_wmma(const __bf16* __restrict__ G,
                                                   const __bf16* __restrict__ W27,
                                                   const float* __restrict__ bias,
                                                   const float* __restrict__ bng,
                                                   const float* __restrict__ bnb,
                                                   float* __restrict__ Yv,
                                                   int O, int C, int r) {
    __shared__ __align__(32) __bf16 sg[2][16 * 128];
    const int r3 = r * r * r;
    const int tid = threadIdx.x;
    const int lane = tid & 31;
    const int wv = tid >> 5;
    const int m0w = blockIdx.y * 64 + wv * 16;
    const int b = blockIdx.z;
    const int vtile = blockIdx.x;
    const int col = lane & 15;
    const int half = lane >> 4;

    v8f acc;
#pragma unroll
    for (int v = 0; v < 8; v++) acc[v] = 0.f;

    stage_g(G, sg[0], b, r, r3, C, 0, vtile, tid);
    __syncthreads();
    int p = 0;
    for (int t = 0; t < 27; t++) {
        if (t + 1 < 27) stage_g(G, sg[p ^ 1], b, r, r3, C, t + 1, vtile, tid);
        const __bf16* wt = W27 + ((size_t)t * O + m0w + col) * C;
        for (int k0 = 0; k0 < C; k0 += 32) {
            v16bf a = *(const v16bf*)(wt + k0 + half * 16);
            v16bf bf = *(const v16bf*)(sg[p] + col * C + k0 + half * 16);
            acc = __builtin_amdgcn_wmma_f32_16x16x32_bf16(
                false, a, false, bf, (short)0, acc, false, false);
        }
        __syncthreads();
        p ^= 1;
    }

    const int vb = vtile * 16 + col;
    float* yr = Yv + ((size_t)b * r3 + vb) * C + m0w + half * 8;
#pragma unroll
    for (int v = 0; v < 8; v++) {
        int m = m0w + half * 8 + v;
        float val = (acc[v] + bias[m]) * (bng[m] * BNINV) + bnb[m];
        yr[v] = (val > 0.f) ? val : 0.1f * val;
    }
}

// ---------------------------------------------------------------------------
// Voxelize: average-scatter points into grid (global_atomic_add_f32)
// ---------------------------------------------------------------------------
__global__ void vox_scatter(const float* __restrict__ f, const float* __restrict__ ncu,
                            float* __restrict__ gsum, float* __restrict__ cnt,
                            int C, int r) {
    int cq = C / 4;
    size_t i = (size_t)blockIdx.x * blockDim.x + threadIdx.x;
    size_t total = (size_t)BNPTS * cq;
    if (i >= total) return;
    int q = (int)(i % cq);
    size_t pn = i / cq;
    int b = (int)(pn / NPTS);
    const float* u = ncu + pn * 3;
    int vi[3];
    for (int k = 0; k < 3; k++) {
        float v = fminf(fmaxf(u[k] * (float)r, 0.f), (float)(r - 1));
        vi[k] = (int)rintf(v);
    }
    int r3 = r * r * r;
    int fl = (vi[0] * r + vi[1]) * r + vi[2];
    float* gp = gsum + ((size_t)b * r3 + fl) * C + q * 4;
    const float* fp = f + pn * (size_t)C + q * 4;
    for (int j = 0; j < 4; j++) atomicAdd(gp + j, fp[j]);
    if (q == 0) atomicAdd(cnt + (size_t)b * r3 + fl, 1.f);
}

__global__ void vox_finalize(float* __restrict__ g, const float* __restrict__ cnt,
                             int C, int r3) {
    size_t i = (size_t)blockIdx.x * blockDim.x + threadIdx.x;
    size_t total = (size_t)BATCH * r3 * C;
    if (i >= total) return;
    float c = cnt[i / C];
    g[i] = g[i] / fmaxf(c, 1.f);
}

// ---------------------------------------------------------------------------
// Devoxelize: trilinear gather
// ---------------------------------------------------------------------------
__global__ void devox_kernel(const float* __restrict__ g, const float* __restrict__ ncu,
                             float* __restrict__ out, int C, int r) {
    int cq = C / 4;
    size_t i = (size_t)blockIdx.x * blockDim.x + threadIdx.x;
    size_t total = (size_t)BNPTS * cq;
    if (i >= total) return;
    int q = (int)(i % cq);
    size_t pn = i / cq;
    int b = (int)(pn / NPTS);
    const float* u = ncu + pn * 3;
    float nc[3], fr[3];
    int x0[3];
    for (int k = 0; k < 3; k++) {
        nc[k] = fminf(fmaxf(u[k] * (float)r, 0.f), (float)(r - 1));
        float fl = floorf(nc[k]);
        x0[k] = (int)fl;
        fr[k] = nc[k] - fl;
    }
    int r3 = r * r * r;
    float a0 = 0.f, a1 = 0.f, a2 = 0.f, a3 = 0.f;
    for (int cc = 0; cc < 8; cc++) {
        int dx = (cc >> 2) & 1, dy = (cc >> 1) & 1, dz = cc & 1;
        int ix = min(x0[0] + dx, r - 1);
        int iy = min(x0[1] + dy, r - 1);
        int iz = min(x0[2] + dz, r - 1);
        float w = (dx ? fr[0] : 1.f - fr[0]) * (dy ? fr[1] : 1.f - fr[1]) *
                  (dz ? fr[2] : 1.f - fr[2]);
        const float* gp = g + ((size_t)b * r3 + ((ix * r + iy) * r + iz)) * C + q * 4;
        a0 += gp[0] * w; a1 += gp[1] * w; a2 += gp[2] * w; a3 += gp[3] * w;
    }
    float* op = out + pn * (size_t)C + q * 4;
    op[0] = a0; op[1] = a1; op[2] = a2; op[3] = a3;
}

// ---------------------------------------------------------------------------
// SE block pieces
// ---------------------------------------------------------------------------
__global__ void se_pool_kernel(const float* __restrict__ g, float* __restrict__ seg,
                               int C, int r3) {
    int b = blockIdx.x / C, c = blockIdx.x % C;
    float s = 0.f;
    for (int v = threadIdx.x; v < r3; v += blockDim.x)
        s += g[((size_t)b * r3 + v) * C + c];
    __shared__ float r[256];
    r[threadIdx.x] = s;
    __syncthreads();
    for (int st = 128; st > 0; st >>= 1) {
        if (threadIdx.x < st) r[threadIdx.x] += r[threadIdx.x + st];
        __syncthreads();
    }
    if (threadIdx.x == 0) seg[blockIdx.x] = r[0] / (float)r3;
}

__global__ void se_fc_kernel(const float* __restrict__ seg,
                             const float* __restrict__ w1, const float* __restrict__ b1,
                             const float* __restrict__ w2, const float* __restrict__ b2,
                             float* __restrict__ ses, int C) {
    int C8 = C / 8;
    __shared__ float h[32];
    for (int b = 0; b < BATCH; b++) {
        if ((int)threadIdx.x < C8) {
            float s = b1[threadIdx.x];
            for (int c = 0; c < C; c++) s += seg[b * C + c] * w1[c * C8 + threadIdx.x];
            h[threadIdx.x] = fmaxf(s, 0.f);
        }
        __syncthreads();
        for (int c = threadIdx.x; c < C; c += blockDim.x) {
            float s = b2[c];
            for (int cc = 0; cc < C8; cc++) s += h[cc] * w2[cc * C + c];
            ses[b * C + c] = sigmoidf_(s);
        }
        __syncthreads();
    }
}

__global__ void se_scale_kernel(float* __restrict__ g, const float* __restrict__ ses,
                                int C, int r3) {
    size_t i = (size_t)blockIdx.x * blockDim.x + threadIdx.x;
    size_t total = (size_t)BATCH * r3 * C;
    if (i >= total) return;
    int c = (int)(i % C);
    int b = (int)(i / ((size_t)r3 * C));
    g[i] *= ses[b * C + c];
}

// ---------------------------------------------------------------------------
// GroupNorm stats (32 groups) over [B][N][C]
// ---------------------------------------------------------------------------
__global__ void gn_stats_kernel(const float* __restrict__ f, float* __restrict__ mu,
                                float* __restrict__ rstd, int C, int Nn) {
    int b = blockIdx.x / 32, gg = blockIdx.x % 32;
    int cg = C / 32;
    float s = 0.f, s2 = 0.f;
    for (int n = threadIdx.x; n < Nn; n += blockDim.x) {
        const float* p = f + ((size_t)b * Nn + n) * C + gg * cg;
        for (int cc = 0; cc < cg; cc++) {
            float v = p[cc];
            s += v; s2 += v * v;
        }
    }
    __shared__ float r1[256], r2[256];
    r1[threadIdx.x] = s; r2[threadIdx.x] = s2;
    __syncthreads();
    for (int st = 128; st > 0; st >>= 1) {
        if (threadIdx.x < st) {
            r1[threadIdx.x] += r1[threadIdx.x + st];
            r2[threadIdx.x] += r2[threadIdx.x + st];
        }
        __syncthreads();
    }
    if (threadIdx.x == 0) {
        float cntv = (float)cg * (float)Nn;
        float m = r1[0] / cntv;
        float v = r2[0] / cntv - m * m;
        mu[blockIdx.x] = m;
        rstd[blockIdx.x] = rsqrtf(v + 1e-5f);
    }
}

// gb[b][2C] = emb[b] @ filmW + filmB
__global__ void film_gb_kernel(const float* __restrict__ emb, const float* __restrict__ w,
                               const float* __restrict__ bb, float* __restrict__ gb, int C2) {
    int i = blockIdx.x * blockDim.x + threadIdx.x;
    if (i >= BATCH * C2) return;
    int b = i / C2, oc = i % C2;
    float s = bb[oc];
    for (int j = 0; j < EMB; j++) s += emb[b * EMB + j] * w[j * C2 + oc];
    gb[i] = s;
}

// out = f + (GN(f)*(1+g) + beta)
__global__ void film_add_kernel(const float* __restrict__ f, const float* __restrict__ mu,
                                const float* __restrict__ rstd, const float* __restrict__ gng,
                                const float* __restrict__ gnb, const float* __restrict__ gb,
                                float* __restrict__ dst, int C) {
    size_t i = (size_t)blockIdx.x * blockDim.x + threadIdx.x;
    size_t total = (size_t)BNPTS * C;
    if (i >= total) return;
    int c = (int)(i % C);
    size_t pn = i / C;
    int b = (int)(pn / NPTS);
    int gg = c / (C / 32);
    float y = (f[i] - mu[b * 32 + gg]) * rstd[b * 32 + gg] * gng[c] + gnb[c];
    dst[i] = f[i] + y * (1.f + gb[(size_t)b * 2 * C + c]) + gb[(size_t)b * 2 * C + C + c];
}

// head: bf16(silu(GN(f)))
__global__ void gn_silu_bf16_kernel(const float* __restrict__ f, const float* __restrict__ mu,
                                    const float* __restrict__ rstd, const float* __restrict__ g,
                                    const float* __restrict__ bb, __bf16* __restrict__ xb, int C) {
    size_t i = (size_t)blockIdx.x * blockDim.x + threadIdx.x;
    size_t total = (size_t)BNPTS * C;
    if (i >= total) return;
    int c = (int)(i % C);
    size_t pn = i / C;
    int b = (int)(pn / NPTS);
    int gg = c / (C / 32);
    float y = (f[i] - mu[b * 32 + gg]) * rstd[b * 32 + gg] * g[c] + bb[c];
    y = y * sigmoidf_(y);
    xb[i] = (__bf16)y;
}

// ---------------------------------------------------------------------------
// Global max pool + gmlp
// ---------------------------------------------------------------------------
__global__ void maxpool_kernel(const float* __restrict__ f, float* __restrict__ gv, int C) {
    int b = blockIdx.x / C, c = blockIdx.x % C;
    float mx = -3.4e38f;
    for (int n = threadIdx.x; n < NPTS; n += blockDim.x)
        mx = fmaxf(mx, f[((size_t)b * NPTS + n) * C + c]);
    __shared__ float r[256];
    r[threadIdx.x] = mx;
    __syncthreads();
    for (int st = 128; st > 0; st >>= 1) {
        if (threadIdx.x < st) r[threadIdx.x] = fmaxf(r[threadIdx.x], r[threadIdx.x + st]);
        __syncthreads();
    }
    if (threadIdx.x == 0) gv[blockIdx.x] = r[0];
}

__global__ void gmlp_kernel(const float* __restrict__ gv,
                            const float* __restrict__ w1, const float* __restrict__ b1,
                            const float* __restrict__ w2, const float* __restrict__ b2,
                            float* __restrict__ g2) {
    __shared__ float h[128];
    for (int b = 0; b < BATCH; b++) {
        for (int o = threadIdx.x; o < 128; o += blockDim.x) {
            float s = b1[o];
            for (int i = 0; i < 128; i++) s += gv[b * 128 + i] * w1[i * 128 + o];
            h[o] = s * sigmoidf_(s);
        }
        __syncthreads();
        for (int o = threadIdx.x; o < 128; o += blockDim.x) {
            float s = b2[o];
            for (int i = 0; i < 128; i++) s += h[i] * w2[i * 128 + o];
            g2[b * 128 + o] = s;
        }
        __syncthreads();
    }
}

// concat [m0(64) | m1(128) | m2(128) | g2 bcast(128)] -> bf16 [bn][448]
__global__ void head_concat_kernel(const float* __restrict__ m0, const float* __restrict__ m1,
                                   const float* __restrict__ m2, const float* __restrict__ g2,
                                   __bf16* __restrict__ xb) {
    size_t i = (size_t)blockIdx.x * blockDim.x + threadIdx.x;
    size_t total = (size_t)BNPTS * 448;
    if (i >= total) return;
    int c = (int)(i % 448);
    size_t pn = i / 448;
    int b = (int)(pn / NPTS);
    float v;
    if (c < 64)       v = m0[pn * 64 + c];
    else if (c < 192) v = m1[pn * 128 + (c - 64)];
    else if (c < 320) v = m2[pn * 128 + (c - 192)];
    else              v = g2[b * 128 + (c - 320)];
    xb[i] = (__bf16)v;
}

// out = alpha*ctx_pv + (1-alpha)*ctx_glb, alpha = sigmoid(10*(t-0.4))
__global__ void blend_kernel(const float* __restrict__ pv, const float* __restrict__ cg,
                             const float* __restrict__ t, float* __restrict__ out) {
    size_t i = (size_t)blockIdx.x * blockDim.x + threadIdx.x;
    size_t total = (size_t)BNPTS * CTX;
    if (i >= total) return;
    int c = (int)(i % CTX);
    size_t pn = i / CTX;
    int b = (int)(pn / NPTS);
    float a = sigmoidf_(10.f * (t[b] - 0.4f));
    out[i] = a * pv[i] + (1.f - a) * cg[b * CTX + c];
}

// ---------------------------------------------------------------------------
// Host orchestration
// ---------------------------------------------------------------------------
struct BlockP {
    const float *bn1_b, *bn1_g, *bn2_b, *bn2_g;
    const float *c1_b, *c1_w, *c2_b, *c2_w;
    const float *film_b, *film_gn_b, *film_gn_g, *film_w;
    const float *pbn_b, *pbn_g, *pmlp_b, *pmlp_w;
    const float *post_bn_b, *post_bn_g, *post_c_b, *post_c_w;
    const float *se_b1, *se_b2, *se_w1, *se_w2;
};
struct StageP {
    BlockP blk[2];
    const float *proj_bn_b, *proj_bn_g, *proj_c_b, *proj_c_w;
};

static inline int cdiv(size_t a, int b) { return (int)((a + b - 1) / b); }

extern "C" void kernel_launch(void* const* d_in, const int* in_sizes, int n_in,
                              void* d_out, int out_size, void* d_ws, size_t ws_size,
                              hipStream_t stream) {
    (void)in_sizes; (void)n_in; (void)out_size; (void)ws_size;

    const float* x    = (const float*)d_in[0];   // (B,N,3)
    const float* t_in = (const float*)d_in[1];   // (B,)
    const float* cond = (const float*)d_in[2];   // (B,COND)

    // ---- params: jax tree flatten order (sorted dict keys, list order) ----
    int pi = 3;
    auto nextp = [&]() { return (const float*)d_in[pi++]; };
    const float* cproj_b = nextp(); const float* cproj_w = nextp();
    const float* ctxe_b  = nextp(); const float* ctxe_w  = nextp();
    const float* l1_b = nextp(); const float* l1_w = nextp();
    const float* l2_b = nextp(); const float* l2_w = nextp();
    const float* hgn_b = nextp(); const float* hgn_g = nextp();
    const float* hout_b = nextp(); const float* hout_w = nextp();
    const float* hpre_b = nextp(); const float* hpre_w = nextp();
    StageP st[3];
    for (int s = 0; s < 3; s++) {
        for (int k = 0; k < 2; k++) {
            BlockP& bp = st[s].blk[k];
            bp.bn1_b = nextp(); bp.bn1_g = nextp();
            bp.bn2_b = nextp(); bp.bn2_g = nextp();
            bp.c1_b = nextp();  bp.c1_w = nextp();
            bp.c2_b = nextp();  bp.c2_w = nextp();
            bp.film_b = nextp(); bp.film_gn_b = nextp();
            bp.film_gn_g = nextp(); bp.film_w = nextp();
            bp.pbn_b = nextp(); bp.pbn_g = nextp();
            bp.pmlp_b = nextp(); bp.pmlp_w = nextp();
            bp.post_bn_b = nextp(); bp.post_bn_g = nextp();
            bp.post_c_b = nextp(); bp.post_c_w = nextp();
            bp.se_b1 = nextp(); bp.se_b2 = nextp();
            bp.se_w1 = nextp(); bp.se_w2 = nextp();
        }
        st[s].proj_bn_b = nextp(); st[s].proj_bn_g = nextp();
        st[s].proj_c_b = nextp();  st[s].proj_c_w = nextp();
    }
    const float* tproj_b = nextp(); const float* tproj_w = nextp();

    // ---- workspace bump allocator ----
    char* ws = (char*)d_ws;
    size_t off = 0;
    auto alloc = [&](size_t bytes) -> void* {
        off = (off + 255) & ~((size_t)255);
        void* p = ws + off;
        off += bytes;
        return p;
    };
    __bf16* xb   = (__bf16*)alloc((size_t)BNPTS * 448 * 2);      // GEMM input staging
    float*  fw1  = (float*)alloc((size_t)BNPTS * 128 * 4);
    float*  fw2  = (float*)alloc((size_t)BNPTS * 128 * 4);
    float*  m0   = (float*)alloc((size_t)BNPTS * 64 * 4);
    float*  m1   = (float*)alloc((size_t)BNPTS * 128 * 4);
    float*  m2   = (float*)alloc((size_t)BNPTS * 128 * 4);
    float*  gridA = (float*)alloc((size_t)BATCH * 32768 * 64 * 4); // max over stages
    float*  gridB = (float*)alloc((size_t)BATCH * 32768 * 64 * 4);
    __bf16* gxb  = (__bf16*)alloc((size_t)BATCH * 32768 * 64 * 2);
    float*  cnt  = (float*)alloc((size_t)BATCH * 32768 * 4);
    float*  ncu  = (float*)alloc((size_t)BNPTS * 3 * 4);
    __bf16* wsc  = (__bf16*)alloc((size_t)27 * 128 * 128 * 2);    // JIT bf16 weights
    float*  embv = (float*)alloc(BATCH * EMB * 4);
    float*  ctxg = (float*)alloc(BATCH * CTX * 4);
    float*  ctr  = (float*)alloc(BATCH * 3 * 4);
    float*  rad  = (float*)alloc(BATCH * 4);
    float*  muv  = (float*)alloc(BATCH * 32 * 4);
    float*  rstd = (float*)alloc(BATCH * 32 * 4);
    float*  gbv  = (float*)alloc(BATCH * 256 * 4);
    float*  seg  = (float*)alloc(BATCH * 128 * 4);
    float*  ses  = (float*)alloc(BATCH * 128 * 4);
    float*  gv   = (float*)alloc(BATCH * 128 * 4);
    float*  g2v  = (float*)alloc(BATCH * 128 * 4);

    const int TB = 256;

    // ---- embeddings & coordinate normalization ----
    emb_kernel<<<BATCH, 256, 0, stream>>>(t_in, cond, tproj_w, tproj_b, cproj_w, cproj_b,
                                          ctxe_w, ctxe_b, embv, ctxg);
    coord_mean_kernel<<<BATCH, 256, 0, stream>>>(x, ctr);
    coord_rad_kernel<<<BATCH, 256, 0, stream>>>(x, ctr, rad);
    ncu_kernel<<<cdiv(BNPTS, TB), TB, 0, stream>>>(x, ctr, rad, ncu);

    const int SC[3] = {64, 128, 128};
    const int SR[3] = {32, 16, 8};
    const int IN_C[3] = {259, 64, 128};
    const int IN_PAD[3] = {288, 64, 128};

    const float* curf = nullptr;
    int curC = 0;

    for (int s = 0; s < 3; s++) {
        const int C = SC[s], r = SR[s], r3 = r * r * r;
        const int Ipad = IN_PAD[s];

        // stage input -> bf16 staging
        if (s == 0) {
            build_f0_kernel<<<cdiv((size_t)BNPTS * 288, TB), TB, 0, stream>>>(embv, x, xb);
        } else {
            to_bf16_kernel<<<cdiv((size_t)BNPTS * curC, TB), TB, 0, stream>>>(
                curf, xb, curC, curC, BNPTS);
        }
        // proj: shared_mlp = relu(bn(conv1x1))
        wconv_kernel<<<cdiv((size_t)C * Ipad, TB), TB, 0, stream>>>(
            st[s].proj_c_w, wsc, C, IN_C[s], Ipad);
        gemm_wmma<1, 1, 0><<<dim3(BNPTS / 64, C / 64), 128, 0, stream>>>(
            xb, wsc, st[s].proj_c_b, st[s].proj_bn_g, st[s].proj_bn_b, nullptr, fw1, C, Ipad);
        curf = fw1; curC = C;

        for (int k = 0; k < 2; k++) {
            const BlockP& bp = st[s].blk[k];
            // --- voxelize ---
            zero_f32<<<cdiv((size_t)BATCH * r3 * C, TB), TB, 0, stream>>>(gridA,
                                                                          (size_t)BATCH * r3 * C);
            zero_f32<<<cdiv((size_t)BATCH * r3, TB), TB, 0, stream>>>(cnt, (size_t)BATCH * r3);
            vox_scatter<<<cdiv((size_t)BNPTS * (C / 4), TB), TB, 0, stream>>>(
                curf, ncu, gridA, cnt, C, r);
            vox_finalize<<<cdiv((size_t)BATCH * r3 * C, TB), TB, 0, stream>>>(gridA, cnt, C, r3);
            // --- conv1 (implicit-GEMM WMMA) + bn + leaky ---
            to_bf16_kernel<<<cdiv((size_t)BATCH * r3 * C, TB), TB, 0, stream>>>(
                gridA, gxb, C, C, (size_t)BATCH * r3);
            wconv3d_kernel<<<cdiv((size_t)27 * C * C, TB), TB, 0, stream>>>(bp.c1_w, wsc, C, C);
            conv3d_wmma<<<dim3(r3 / 16, C / 64, BATCH), 128, 0, stream>>>(
                gxb, wsc, bp.c1_b, bp.bn1_g, bp.bn1_b, gridB, C, C, r);
            // --- conv2 ---
            to_bf16_kernel<<<cdiv((size_t)BATCH * r3 * C, TB), TB, 0, stream>>>(
                gridB, gxb, C, C, (size_t)BATCH * r3);
            wconv3d_kernel<<<cdiv((size_t)27 * C * C, TB), TB, 0, stream>>>(bp.c2_w, wsc, C, C);
            conv3d_wmma<<<dim3(r3 / 16, C / 64, BATCH), 128, 0, stream>>>(
                gxb, wsc, bp.c2_b, bp.bn2_g, bp.bn2_b, gridA, C, C, r);
            // --- SE ---
            se_pool_kernel<<<BATCH * C, 256, 0, stream>>>(gridA, seg, C, r3);
            se_fc_kernel<<<1, 128, 0, stream>>>(seg, bp.se_w1, bp.se_b1, bp.se_w2, bp.se_b2,
                                                ses, C);
            se_scale_kernel<<<cdiv((size_t)BATCH * r3 * C, TB), TB, 0, stream>>>(gridA, ses, C, r3);
            // --- devoxelize -> fw2 ---
            devox_kernel<<<cdiv((size_t)BNPTS * (C / 4), TB), TB, 0, stream>>>(
                gridA, ncu, fw2, C, r);
            // --- point branch: relu(bn(pmlp(f))) + vf -> fw1 ---
            to_bf16_kernel<<<cdiv((size_t)BNPTS * C, TB), TB, 0, stream>>>(curf, xb, C, C, BNPTS);
            wconv_kernel<<<cdiv((size_t)C * C, TB), TB, 0, stream>>>(bp.pmlp_w, wsc, C, C, C);
            gemm_wmma<1, 1, 1><<<dim3(BNPTS / 64, C / 64), 128, 0, stream>>>(
                xb, wsc, bp.pmlp_b, bp.pbn_g, bp.pbn_b, fw2, fw1, C, C);
            // --- post shared_mlp -> fw2 ---
            to_bf16_kernel<<<cdiv((size_t)BNPTS * C, TB), TB, 0, stream>>>(fw1, xb, C, C, BNPTS);
            wconv_kernel<<<cdiv((size_t)C * C, TB), TB, 0, stream>>>(bp.post_c_w, wsc, C, C, C);
            gemm_wmma<1, 1, 0><<<dim3(BNPTS / 64, C / 64), 128, 0, stream>>>(
                xb, wsc, bp.post_c_b, bp.post_bn_g, bp.post_bn_b, nullptr, fw2, C, C);
            // --- FiLM: out = f2 + (GN(f2)*(1+g)+beta) ---
            gn_stats_kernel<<<BATCH * 32, 256, 0, stream>>>(fw2, muv, rstd, C, NPTS);
            film_gb_kernel<<<cdiv((size_t)BATCH * 2 * C, TB), TB, 0, stream>>>(
                embv, bp.film_w, bp.film_b, gbv, 2 * C);
            float* dst = (k == 1) ? ((s == 0) ? m0 : (s == 1) ? m1 : m2) : fw1;
            film_add_kernel<<<cdiv((size_t)BNPTS * C, TB), TB, 0, stream>>>(
                fw2, muv, rstd, bp.film_gn_g, bp.film_gn_b, gbv, dst, C);
            curf = dst;
        }
    }

    // ---- global branch ----
    maxpool_kernel<<<BATCH * 128, 256, 0, stream>>>(m2, gv, 128);
    gmlp_kernel<<<1, 128, 0, stream>>>(gv, l1_w, l1_b, l2_w, l2_b, g2v);

    // ---- head ----
    head_concat_kernel<<<cdiv((size_t)BNPTS * 448, TB), TB, 0, stream>>>(m0, m1, m2, g2v, xb);
    wconv_kernel<<<cdiv((size_t)128 * 448, TB), TB, 0, stream>>>(hpre_w, wsc, 128, 448, 448);
    gemm_wmma<0, 0, 0><<<dim3(BNPTS / 64, 128 / 64), 128, 0, stream>>>(
        xb, wsc, hpre_b, nullptr, nullptr, nullptr, fw1, 128, 448);
    gn_stats_kernel<<<BATCH * 32, 256, 0, stream>>>(fw1, muv, rstd, 128, NPTS);
    gn_silu_bf16_kernel<<<cdiv((size_t)BNPTS * 128, TB), TB, 0, stream>>>(
        fw1, muv, rstd, hgn_g, hgn_b, xb, 128);
    wconv_kernel<<<cdiv((size_t)64 * 128, TB), TB, 0, stream>>>(hout_w, wsc, 64, 128, 128);
    gemm_wmma<0, 0, 0><<<dim3(BNPTS / 64, 64 / 64), 128, 0, stream>>>(
        xb, wsc, hout_b, nullptr, nullptr, nullptr, fw2, 64, 128);

    // ---- time-gated blend into d_out (B,N,CTX) fp32 ----
    blend_kernel<<<cdiv((size_t)BNPTS * CTX, TB), TB, 0, stream>>>(
        fw2, ctxg, t_in, (float*)d_out);
}
